// SelfAttention2d_609885356738
// MI455X (gfx1250) — matrix operand order repaired and checked by reading, not compiled, not run on hardware
//
#include <hip/hip_runtime.h>
#include <hip/hip_bf16.h>

// ---------------------------------------------------------------------------
// SelfAttention2d for MI455X (gfx1250): GroupNorm -> QKV(1x1) -> 4-head
// flash attention over 4096 positions -> proj(1x1) -> +residual.
// GEMMs: v_wmma_f32_16x16x32_f16 (fp32 accumulate), wave32.
// Attention stages K/V chunks into LDS with async global->LDS copies
// (ASYNCcnt / s_wait_asynccnt), double-buffered, shared by 4 waves.
// ---------------------------------------------------------------------------

typedef __attribute__((ext_vector_type(16))) _Float16 v16h;
typedef __attribute__((ext_vector_type(8)))  _Float16 v8h;
typedef __attribute__((ext_vector_type(8)))  float    v8f;
typedef __attribute__((ext_vector_type(4)))  int      v4i;

#define CCH 256      // channels
#define NSP 4096     // spatial positions (64*64)
#define NHH 4        // heads
#define HDD 64       // head dim
#define NBB 2        // batch
#define QKV_O 768    // 3*C
#define KCH 64       // keys per attention chunk
#define KLD 72       // padded LDS row (halves): 144B rows -> conflict-free b128

#if defined(__gfx1250__) && defined(__has_builtin)
#if __has_builtin(__builtin_amdgcn_global_load_async_to_lds_b128)
#define USE_ASYNC_LDS 1
#endif
#endif
#ifndef USE_ASYNC_LDS
#define USE_ASYNC_LDS 0
#endif

__device__ __forceinline__ v8f wmma16(v16h a, v16h b, v8f c) {
  // D = A(16x32 f16) * B(32x16 f16) + C(16x16 f32)
  return __builtin_amdgcn_wmma_f32_16x16x32_f16(
      /*neg_a=*/false, a, /*neg_b=*/false, b,
      /*c_mod=*/(short)0, c, /*reuse_a=*/false, /*reuse_b=*/false);
}

// A-matrix 16x32 f16 fragment (ISA 7.12.2): lane holds row M=lane&15,
// two contiguous 8-half runs at K = 8*half and K = 16 + 8*half.
__device__ __forceinline__ v16h load_a16(const _Float16* tile, int ld, int k0, int lane) {
  const int m = lane & 15, hh = lane >> 4;
  const _Float16* p = tile + (size_t)m * ld + k0 + 8 * hh;
  v8h lo = *(const v8h*)(p);
  v8h hi = *(const v8h*)(p + 16);
  return __builtin_shufflevector(lo, hi, 0,1,2,3,4,5,6,7,8,9,10,11,12,13,14,15);
}

// B-matrix 32x16 f16 fragment: lane holds column N=lane&15, two contiguous
// 8-half runs at K = 8*half and K = 16 + 8*half (K-contiguous memory per column).
__device__ __forceinline__ v16h load_b16(const _Float16* tile, int ld, int k0, int lane) {
  const int n = lane & 15, hh = lane >> 4;
  const _Float16* p = tile + (size_t)n * ld + k0 + 8 * hh;
  v8h lo = *(const v8h*)(p);
  v8h hi = *(const v8h*)(p + 16);
  return __builtin_shufflevector(lo, hi, 0,1,2,3,4,5,6,7,8,9,10,11,12,13,14,15);
}

// Wave-private LDS producer->consumer ordering (CDNA5 split counters).
__device__ __forceinline__ void lds_fence() {
  asm volatile("s_wait_dscnt 0" ::: "memory");
}

__device__ __forceinline__ void async_wait0() {
#if USE_ASYNC_LDS
  asm volatile("s_wait_asynccnt 0" ::: "memory");
#endif
}

// One 16-byte global->LDS transfer per lane.
__device__ __forceinline__ void copy_b128(const _Float16* g, _Float16* l) {
#if USE_ASYNC_LDS
  void* gp = (void*)g;   // drop const, then explicit address-space casts
  void* lp = (void*)l;
  __builtin_amdgcn_global_load_async_to_lds_b128(
      (__attribute__((address_space(1))) v4i*)gp,
      (__attribute__((address_space(3))) v4i*)lp, 0, 0);
#else
  *(v8h*)l = *(const v8h*)g;
#endif
}

// ---------------------------------------------------------------------------
// Kernel 0: fp32 -> f16 weight conversion (qkv_w 768x256, proj_w 256x256)
// ---------------------------------------------------------------------------
__global__ void cvt_weights_kernel(const float* __restrict__ wq,
                                   const float* __restrict__ wp,
                                   _Float16* __restrict__ wq_h,
                                   _Float16* __restrict__ wp_h) {
  int i = blockIdx.x * 256 + threadIdx.x;
  if (i < QKV_O * CCH) wq_h[i] = (_Float16)wq[i];
  if (i < CCH * CCH)   wp_h[i] = (_Float16)wp[i];
}

// ---------------------------------------------------------------------------
// Kernel 1: GroupNorm (32 groups of 8 ch) + transpose to xn_t[b][n][c] f16
// One 256-thread block per (b, group): 8*4096 = 32768 elements.
// ---------------------------------------------------------------------------
__global__ void __launch_bounds__(256) gn_kernel(const float* __restrict__ x,
                                                 const float* __restrict__ gamma,
                                                 const float* __restrict__ beta,
                                                 _Float16* __restrict__ xnt) {
  const int bg = blockIdx.x;
  const int b  = bg >> 5;
  const int g  = bg & 31;
  const int tid = threadIdx.x;
  const float* xp = x + ((size_t)b * CCH + g * 8) * NSP;

  float s = 0.f, s2 = 0.f;
  for (int i = tid; i < 8 * NSP; i += 256) {
    float v = xp[i];
    s += v; s2 += v * v;
  }
  __shared__ float rs[256], rs2[256];
  rs[tid] = s; rs2[tid] = s2;
  __syncthreads();
  for (int st = 128; st > 0; st >>= 1) {
    if (tid < st) { rs[tid] += rs[tid + st]; rs2[tid] += rs2[tid + st]; }
    __syncthreads();
  }
  const float inv_n = 1.0f / (8.0f * NSP);
  const float mean = rs[0] * inv_n;
  const float var  = rs2[0] * inv_n - mean * mean;
  const float rstd = rsqrtf(var + 1e-5f);

  for (int i = tid; i < 8 * NSP; i += 256) {
    int c  = i >> 12;         // 0..7
    int n  = i & (NSP - 1);
    int ch = g * 8 + c;
    float v = (xp[i] - mean) * rstd * gamma[ch] + beta[ch];
    xnt[((size_t)b * NSP + n) * CCH + ch] = (_Float16)v;
  }
}

// ---------------------------------------------------------------------------
// Kernel 2: QKV GEMM: (768x256) x (256 x 4096) per batch, bias, scatter into
// q,k: [bh][n][64] (q pre-scaled by 1/8), v: [bh][64][n].
// One wave computes a 16(o) x 64(n) strip. 6144 waves.
// ---------------------------------------------------------------------------
__global__ void __launch_bounds__(128) qkv_kernel(const _Float16* __restrict__ wq_h,
                                                  const float* __restrict__ qkvb,
                                                  const _Float16* __restrict__ xnt,
                                                  _Float16* __restrict__ qb,
                                                  _Float16* __restrict__ kb,
                                                  _Float16* __restrict__ vb) {
  const int lane = threadIdx.x & 31;
  const int wv   = threadIdx.x >> 5;
  const int wid  = blockIdx.x * 4 + wv;          // 0..6143 = 2 * 48 * 64
  const int nstrip = wid & 63;
  const int t2 = wid >> 6;
  const int ot = t2 % 48;
  const int b  = t2 / 48;
  const int o0 = ot * 16;
  const int n0 = nstrip * 64;

  const _Float16* A  = wq_h + (size_t)o0 * CCH;                 // rows o, K=c
  const _Float16* Bx = xnt + ((size_t)b * NSP + n0) * CCH;      // cols n, K=c

  v8f d[4] = {};
  #pragma unroll
  for (int k0 = 0; k0 < CCH; k0 += 32) {
    v16h a = load_a16(A, CCH, k0, lane);
    #pragma unroll
    for (int j = 0; j < 4; ++j) {
      v16h bm = load_b16(Bx + (size_t)(16 * j) * CCH, CCH, k0, lane);
      d[j] = wmma16(a, bm, d[j]);
    }
  }

  const int hh = lane >> 4, l15 = lane & 15;
  const int sec    = o0 >> 8;           // 0=q 1=k 2=v
  const int olocal = o0 & 255;
  const int head   = olocal >> 6;
  const int dbase  = (olocal & 63) + 8 * hh;
  const size_t bh  = (size_t)(b * NHH + head);

  float biasv[8];
  #pragma unroll
  for (int r = 0; r < 8; ++r) biasv[r] = qkvb[o0 + 8 * hh + r];

  #pragma unroll
  for (int j = 0; j < 4; ++j) {
    const int n = n0 + 16 * j + l15;
    if (sec == 2) {                         // V: [bh][d][n] (strided once)
      #pragma unroll
      for (int r = 0; r < 8; ++r)
        vb[(bh * HDD + dbase + r) * NSP + n] = (_Float16)(d[j][r] + biasv[r]);
    } else {
      const float sc = (sec == 0) ? 0.125f : 1.0f;   // q * 1/sqrt(64)
      v8h pk;
      #pragma unroll
      for (int r = 0; r < 8; ++r) pk[r] = (_Float16)((d[j][r] + biasv[r]) * sc);
      _Float16* dst = (sec == 0) ? qb : kb;
      *(v8h*)(dst + (bh * NSP + n) * HDD + dbase) = pk;   // contiguous 16B
    }
  }
}

// ---------------------------------------------------------------------------
// Kernel 3: flash attention. Block = 4 waves on 4 adjacent query tiles of the
// SAME (b,h). K/V chunks (64 keys) staged into padded LDS via async
// global->LDS b128 copies, double-buffered; copy of chunk i+1 overlaps the
// WMMAs of chunk i. Online softmax through wave-private LDS.
// ---------------------------------------------------------------------------
__global__ void __launch_bounds__(128) attn_kernel(const _Float16* __restrict__ qb,
                                                   const _Float16* __restrict__ kb,
                                                   const _Float16* __restrict__ vb,
                                                   _Float16* __restrict__ ao) {
  __shared__ __align__(16) _Float16 sK[2][KCH][KLD];   // [key][d], padded
  __shared__ __align__(16) _Float16 sV[2][KCH][KLD];   // [d][key], padded
  __shared__ __align__(32) float    sS[4][16][64];
  __shared__ __align__(32) _Float16 sP[4][16][64];
  __shared__ float sC[4][16];
  __shared__ float sL[4][16];

  const int tid  = threadIdx.x;
  const int lane = tid & 31;
  const int wv   = tid >> 5;
  const int bh   = blockIdx.x >> 6;               // 0..7
  const int qt   = (blockIdx.x & 63) * 4 + wv;    // 0..255
  const int b    = bh >> 2;
  const int hidx = bh & 3;
  const int hh = lane >> 4, l15 = lane & 15;

  const _Float16* Q  = qb + ((size_t)bh * NSP + qt * 16) * HDD; // [q][d]
  const _Float16* Kg = kb + (size_t)bh * NSP * HDD;             // [key][d]
  const _Float16* Vg = vb + (size_t)bh * HDD * NSP;             // [d][key]

  const v16h aq0 = load_a16(Q, HDD, 0,  lane);
  const v16h aq1 = load_a16(Q, HDD, 32, lane);

  v8f o[4] = {};
  float m_i = -1e30f, l_i = 0.f;                  // meaningful in lanes 0..15

  const int nchunks = NSP / KCH;                  // 64

  // Cooperative stage of one 64-key K/V chunk: 8 b128 transfers per wave.
  auto stage = [&](int kc, int buf) {
    #pragma unroll
    for (int i = 0; i < 4; ++i) {
      int e   = i * 128 + tid;                    // 0..511
      int row = e >> 3;                           // 0..63
      int c8  = (e & 7) * 8;                      // half offset 0..56
      copy_b128(Kg + (size_t)(kc + row) * HDD + c8, &sK[buf][row][c8]);
      copy_b128(Vg + (size_t)row * NSP + kc + c8,   &sV[buf][row][c8]);
    }
  };

  stage(0, 0);                                    // prologue

  for (int ic = 0; ic < nchunks; ++ic) {
    async_wait0();          // this wave's share of chunk ic has landed
    __syncthreads();        // whole chunk visible; prev compute done -> other
                            // buffer is free to overwrite
    if (ic + 1 < nchunks) {
      stage((ic + 1) * KCH, (ic + 1) & 1);
      if (ic + 2 < nchunks) {                     // pull chunk ic+2 toward L2
        __builtin_prefetch(Kg + (size_t)(ic + 2) * KCH * HDD, 0, 1);
        __builtin_prefetch(Vg + (size_t)(ic + 2) * KCH, 0, 1);
      }
    }
    const _Float16 (*cK)[KLD] = sK[ic & 1];
    const _Float16 (*cV)[KLD] = sV[ic & 1];

    // ---- S = (Q*scale)^T K : 4 n-tiles x (K=64 = 2 wmma) from LDS ----
    #pragma unroll
    for (int jt = 0; jt < 4; ++jt) {
      v8f s = {};
      s = wmma16(aq0, load_b16(&cK[16 * jt][0], KLD, 0,  lane), s);
      s = wmma16(aq1, load_b16(&cK[16 * jt][0], KLD, 32, lane), s);
      #pragma unroll
      for (int r = 0; r < 8; ++r)
        sS[wv][8 * hh + r][16 * jt + l15] = s[r];
    }
    lds_fence();

    // ---- online softmax: lane<16 owns one query row of 64 scores ----
    if (lane < 16) {
      float mc = m_i;
      #pragma unroll 8
      for (int c = 0; c < 64; ++c) mc = fmaxf(mc, sS[wv][lane][c]);
      const float corr = __expf(m_i - mc);
      float sum = 0.f;
      #pragma unroll 8
      for (int c = 0; c < 64; ++c) {
        float p = __expf(sS[wv][lane][c] - mc);
        sum += p;
        sP[wv][lane][c] = (_Float16)p;
      }
      l_i = l_i * corr + sum;
      m_i = mc;
      sC[wv][lane] = corr;
    }
    lds_fence();

    // ---- rescale O rows, then O += P @ V^T (V from LDS) ----
    float cf[8];
    #pragma unroll
    for (int r = 0; r < 8; ++r) cf[r] = sC[wv][8 * hh + r];
    #pragma unroll
    for (int j = 0; j < 4; ++j)
      #pragma unroll
      for (int r = 0; r < 8; ++r) o[j][r] *= cf[r];

    const v16h ap0 = load_a16(&sP[wv][0][0], 64, 0,  lane);
    const v16h ap1 = load_a16(&sP[wv][0][0], 64, 32, lane);
    #pragma unroll
    for (int dj = 0; dj < 4; ++dj) {
      o[dj] = wmma16(ap0, load_b16(&cV[16 * dj][0], KLD, 0,  lane), o[dj]);
      o[dj] = wmma16(ap1, load_b16(&cV[16 * dj][0], KLD, 32, lane), o[dj]);
    }
  }

  // ---- epilogue: O /= l, write ao[b][n][c] ----
  if (lane < 16) sL[wv][lane] = l_i;
  lds_fence();
  float inv[8];
  #pragma unroll
  for (int r = 0; r < 8; ++r) inv[r] = 1.0f / sL[wv][8 * hh + r];

  #pragma unroll
  for (int dj = 0; dj < 4; ++dj) {
    const int c = hidx * HDD + 16 * dj + l15;
    #pragma unroll
    for (int r = 0; r < 8; ++r) {
      const int n = qt * 16 + 8 * hh + r;
      ao[((size_t)b * NSP + n) * CCH + c] = (_Float16)(o[dj][r] * inv[r]);
    }
  }
}

// ---------------------------------------------------------------------------
// Kernel 4: proj GEMM (256x256) x (256x4096) + bias + residual -> fp32 out
// One wave computes a 16(o) x 64(n) strip. 2048 waves.
// ---------------------------------------------------------------------------
__global__ void __launch_bounds__(128) proj_kernel(const _Float16* __restrict__ wp_h,
                                                   const float* __restrict__ projb,
                                                   const _Float16* __restrict__ ao,
                                                   const float* __restrict__ x,
                                                   float* __restrict__ out) {
  const int lane = threadIdx.x & 31;
  const int wv   = threadIdx.x >> 5;
  const int wid  = blockIdx.x * 4 + wv;        // 0..2047 = 2 * 16 * 64
  const int nstrip = wid & 63;
  const int t2 = wid >> 6;
  const int ot = t2 & 15;
  const int b  = t2 >> 4;
  const int o0 = ot * 16;
  const int n0 = nstrip * 64;

  const _Float16* A  = wp_h + (size_t)o0 * CCH;
  const _Float16* Bx = ao + ((size_t)b * NSP + n0) * CCH;

  v8f d[4] = {};
  #pragma unroll
  for (int k0 = 0; k0 < CCH; k0 += 32) {
    v16h a = load_a16(A, CCH, k0, lane);
    #pragma unroll
    for (int j = 0; j < 4; ++j) {
      v16h bm = load_b16(Bx + (size_t)(16 * j) * CCH, CCH, k0, lane);
      d[j] = wmma16(a, bm, d[j]);
    }
  }

  const int hh = lane >> 4, l15 = lane & 15;
  float biasv[8];
  #pragma unroll
  for (int r = 0; r < 8; ++r) biasv[r] = projb[o0 + 8 * hh + r];

  #pragma unroll
  for (int j = 0; j < 4; ++j) {
    const int n = n0 + 16 * j + l15;
    #pragma unroll
    for (int r = 0; r < 8; ++r) {
      const size_t idx = ((size_t)b * CCH + o0 + 8 * hh + r) * NSP + n;
      out[idx] = d[j][r] + biasv[r] + x[idx];
    }
  }
}

// ---------------------------------------------------------------------------
// Host-side launch. Inputs: x, gamma, beta, qkv_w, qkv_b, proj_w, proj_b.
// ---------------------------------------------------------------------------
extern "C" void kernel_launch(void* const* d_in, const int* in_sizes, int n_in,
                              void* d_out, int out_size, void* d_ws, size_t ws_size,
                              hipStream_t stream) {
  const float* x      = (const float*)d_in[0];
  const float* gamma  = (const float*)d_in[1];
  const float* beta   = (const float*)d_in[2];
  const float* qkv_w  = (const float*)d_in[3];
  const float* qkv_b  = (const float*)d_in[4];
  const float* proj_w = (const float*)d_in[5];
  const float* proj_b = (const float*)d_in[6];
  float* out = (float*)d_out;

  // Workspace layout (bytes, 256-aligned): ~20.5 MB total.
  char* ws = (char*)d_ws;
  _Float16* wq_h = (_Float16*)(ws + 0);                    // 768*256*2 = 384K
  _Float16* wp_h = (_Float16*)(ws + 393216);               // 256*256*2 = 128K
  _Float16* xnt  = (_Float16*)(ws + 524288);               // 2*4096*256*2 = 4M
  _Float16* qb   = (_Float16*)(ws + 4718592);              // 4M
  _Float16* kb   = (_Float16*)(ws + 8912896);              // 4M
  _Float16* vb   = (_Float16*)(ws + 13107200);             // 4M
  _Float16* ao   = (_Float16*)(ws + 17301504);             // 4M

  cvt_weights_kernel<<<768, 256, 0, stream>>>(qkv_w, proj_w, wq_h, wp_h);
  gn_kernel<<<NBB * 32, 256, 0, stream>>>(x, gamma, beta, xnt);
  qkv_kernel<<<1536, 128, 0, stream>>>(wq_h, qkv_b, xnt, qb, kb, vb);
  attn_kernel<<<512, 128, 0, stream>>>(qb, kb, vb, ao);
  proj_kernel<<<512, 128, 0, stream>>>(wp_h, proj_b, ao, x, out);
}
